// FSSGruBackBone_73710228734009
// MI455X (gfx1250) — compile-verified, matrix operand order checked by reading, not run on hardware
//
#include <hip/hip_runtime.h>
#include <hip/hip_bf16.h>
#include <stddef.h>

// ---------------------------------------------------------------------------
// FSSGruBackBone for MI455X (gfx1250, wave32, WMMA).
// All GEMM operands are bf16 (weights pre-converted once per call; activations
// kept as bf16 shadow copies), f32 accumulation via v_wmma_f32_16x16x32_bf16.
// Inner loop is software-pipelined with two explicit chunk buffers so WMMAs on
// chunk k overlap the 10 in-flight loads of chunk k+1 (partial s_wait_loadcnt).
// ---------------------------------------------------------------------------

#define BB   64
#define SS   256
#define HH   1024
#define LL   4
#define GHH  512
#define H3   3072
#define H2   2048

typedef __attribute__((ext_vector_type(16))) __bf16        v16bf;
typedef __attribute__((ext_vector_type(8)))  float         v8f;
typedef __attribute__((ext_vector_type(8)))  unsigned int  v8u;

__device__ __forceinline__ unsigned short f2bf(float f) {
  unsigned int u = __float_as_uint(f);
  return (unsigned short)((u + 0x7FFFu + ((u >> 16) & 1u)) >> 16);   // RNE
}

// ---------------------------------------------------------------------------
// fp32 -> bf16 conversion (weights + x input; L2-resident thereafter)
// ---------------------------------------------------------------------------
__global__ void f32_to_bf16_kernel(const float* __restrict__ src,
                                   unsigned short* __restrict__ dst, int n) {
  int i = blockIdx.x * blockDim.x + threadIdx.x;
  if (i < n) dst[i] = f2bf(src[i]);
}

// h[l,b,:] = hidden0[l,0,:]  (f32 state + bf16 shadow)
__global__ void init_h_kernel(const float* __restrict__ h0,
                              float* __restrict__ h,
                              unsigned short* __restrict__ hbf) {
  int idx = blockIdx.x * blockDim.x + threadIdx.x;   // L*B*H
  int l = idx >> 16;            // B*H = 65536
  int j = idx & (HH - 1);
  float v = h0[(l << 10) + j];
  h[idx] = v;
  hbf[idx] = f2bf(v);
}

// ---------------------------------------------------------------------------
// WMMA GEMM:  C[M,N] = epilogue( A[M,K](bf16) * Bw[N,K](bf16)^T )
//   MODE 0: + bias[col] (bias may be null)    -> writes C (f32)
//   MODE 1: relu                              -> writes Cbf (bf16)
//   MODE 2: o = gate * sigmoid(acc)           -> writes C + Cbf (+ out2)
// One wave computes a 16x64 strip (4 x 16x16 WMMA tiles), K in steps of 32.
// Fragment layout per CDNA5 ISA (16-bit A 16x32 / B 32x16 tables):
//   lane L: row = L&15, K chunks [K0..K0+7] and [K0+16..K0+23], K0=kk+8*(L>>4)
//   C/D: VGPR i -> row = i + 8*(L>>4), col = L&15.
// Requires K % 64 == 0 (even number of 32-wide chunks) -- true for all uses.
// __launch_bounds__(128, 1): allow high VGPR use; per-wave ILP beats occupancy
// for these small latency-bound GEMMs.
// ---------------------------------------------------------------------------
template <int MODE, bool WC, bool WBF>
__global__ __launch_bounds__(128, 1) void wmma_gemm_kernel(
    const unsigned short* __restrict__ A, int lda,
    const unsigned short* __restrict__ Bw,
    const float* __restrict__ bias,
    float* __restrict__ C, int ldc,
    unsigned short* __restrict__ Cbf,
    const float* __restrict__ gate, int ldg,
    float* __restrict__ out2, int ld2,
    int M, int N, int K) {
  int wave   = (int)((blockIdx.x * blockDim.x + threadIdx.x) >> 5);
  int lane   = threadIdx.x & 31;
  int tilesN = N >> 6;
  int total  = (M >> 4) * tilesN;
  if (wave >= total) return;                 // wave-uniform: EXEC stays all-1s
  int mt = (wave / tilesN) << 4;
  int nt = (wave % tilesN) << 6;
  int hl = lane >> 4;                        // half: selects K sub-bank
  int r  = lane & 15;                        // row of A / row of W (=col of B)

  const unsigned short* arow  = A  + (size_t)(mt + r) * (size_t)lda + (hl << 3);
  const unsigned short* brow0 = Bw + (size_t)(nt +  0 + r) * (size_t)K + (hl << 3);
  const unsigned short* brow1 = Bw + (size_t)(nt + 16 + r) * (size_t)K + (hl << 3);
  const unsigned short* brow2 = Bw + (size_t)(nt + 32 + r) * (size_t)K + (hl << 3);
  const unsigned short* brow3 = Bw + (size_t)(nt + 48 + r) * (size_t)K + (hl << 3);

  v8f acc[4] = {v8f{}, v8f{}, v8f{}, v8f{}};

  // load one 32-wide K-chunk (A: 2x b128, B: 8x b128) into a named buffer
  auto loadc = [&](int kk, uint4& a0, uint4& a1, uint4* b) {
    a0   = *(const uint4*)(arow  + kk);
    a1   = *(const uint4*)(arow  + kk + 16);
    b[0] = *(const uint4*)(brow0 + kk);
    b[1] = *(const uint4*)(brow0 + kk + 16);
    b[2] = *(const uint4*)(brow1 + kk);
    b[3] = *(const uint4*)(brow1 + kk + 16);
    b[4] = *(const uint4*)(brow2 + kk);
    b[5] = *(const uint4*)(brow2 + kk + 16);
    b[6] = *(const uint4*)(brow3 + kk);
    b[7] = *(const uint4*)(brow3 + kk + 16);
  };
  // 4 WMMAs on one chunk buffer
  auto mmac = [&](const uint4& a0, const uint4& a1, const uint4* b) {
    v8u apk = {a0.x, a0.y, a0.z, a0.w, a1.x, a1.y, a1.z, a1.w};
    v16bf af = __builtin_bit_cast(v16bf, apk);
#pragma unroll
    for (int s = 0; s < 4; ++s) {
      v8u bpk = {b[2 * s].x, b[2 * s].y, b[2 * s].z, b[2 * s].w,
                 b[2 * s + 1].x, b[2 * s + 1].y, b[2 * s + 1].z,
                 b[2 * s + 1].w};
      v16bf bf = __builtin_bit_cast(v16bf, bpk);
      acc[s] = __builtin_amdgcn_wmma_f32_16x16x32_bf16(
          /*neg_a=*/false, af, /*neg_b=*/false, bf,
          /*c_mod=*/(short)0, acc[s], /*reuse_a=*/false, /*reuse_b=*/false);
    }
  };

  // ---- 2-stage software pipeline over K (chunk count is even)
  uint4 xa0, xa1, xb[8], ya0, ya1, yb[8];
  loadc(0, xa0, xa1, xb);
  int kk = 0;
  for (; kk < K - 64; kk += 64) {
    loadc(kk + 32, ya0, ya1, yb);   // next chunk in flight
    mmac(xa0, xa1, xb);             // compute current
    loadc(kk + 64, xa0, xa1, xb);
    mmac(ya0, ya1, yb);
  }
  loadc(K - 32, ya0, ya1, yb);
  mmac(xa0, xa1, xb);
  mmac(ya0, ya1, yb);

  // ---- straight-line specialized epilogue
#pragma unroll
  for (int s = 0; s < 4; ++s) {
    int col = nt + (s << 4) + r;
#pragma unroll
    for (int i = 0; i < 8; ++i) {
      int row = mt + (hl << 3) + i;
      float v = acc[s][i];
      if constexpr (MODE == 0) {
        if (bias) v += bias[col];
      } else if constexpr (MODE == 1) {
        v = v > 0.f ? v : 0.f;
      } else {
        float g = 1.f / (1.f + __expf(-v));
        v = gate[(size_t)row * ldg + col] * g;
      }
      if constexpr (WC)  C[(size_t)row * ldc + col]   = v;
      if constexpr (WBF) Cbf[(size_t)row * ldc + col] = f2bf(v);
      if constexpr (MODE == 2) {
        if (out2) out2[(size_t)row * ld2 + col] = v;
      }
    }
  }
}

// ---------------------------------------------------------------------------
// GRU pointwise: h' from gi, gh, h (all f32)
// ---------------------------------------------------------------------------
__global__ void gru_pointwise_kernel(const float* __restrict__ gi,
                                     const float* __restrict__ gh,
                                     const float* __restrict__ h,
                                     float* __restrict__ hp) {
  int idx = blockIdx.x * blockDim.x + threadIdx.x;   // B*H
  int b = idx >> 10;
  int j = idx & (HH - 1);
  const float* gib = gi + (size_t)b * H3;
  const float* ghb = gh + (size_t)b * H3;
  float r = 1.f / (1.f + __expf(-(gib[j] + ghb[j])));
  float z = 1.f / (1.f + __expf(-(gib[HH + j] + ghb[HH + j])));
  float n = tanhf(gib[2 * HH + j] + r * ghb[2 * HH + j]);
  hp[idx] = (1.f - z) * n + z * h[idx];
}

// ---------------------------------------------------------------------------
// se_bf[b,0,:] = bf16(h'[b,:]); se_bf[b,1,:] = bf16(sort(h'[b,:]))
// (bitonic in LDS, one row per 512-thread block)
// ---------------------------------------------------------------------------
__global__ __launch_bounds__(512) void sort_rows_kernel(
    const float* __restrict__ hp, unsigned short* __restrict__ se_bf) {
  __shared__ float s[HH];
  int b = blockIdx.x;
  const float* row = hp + ((size_t)b << 10);
  unsigned short* seb = se_bf + ((size_t)b << 11);
  for (int i = threadIdx.x; i < HH; i += 512) {
    float v = row[i];
    s[i] = v;
    seb[i] = f2bf(v);
  }
  for (int k = 2; k <= HH; k <<= 1) {
    for (int j = k >> 1; j > 0; j >>= 1) {
      __syncthreads();
      for (int i = threadIdx.x; i < HH; i += 512) {
        int ixj = i ^ j;
        if (ixj > i) {
          float a = s[i], c = s[ixj];
          bool up = ((i & k) == 0);
          if ((a > c) == up) { s[i] = c; s[ixj] = a; }
        }
      }
    }
  }
  __syncthreads();
  for (int i = threadIdx.x; i < HH; i += 512) seb[HH + i] = f2bf(s[i]);
}

// _shuffle: dst_bf[b,j] = bf16(src[b, (j&3)*512 + (j>>2)])  on (B,2048)
__global__ void shuffle_kernel(const float* __restrict__ src,
                               unsigned short* __restrict__ dst_bf) {
  int idx = blockIdx.x * blockDim.x + threadIdx.x;   // B*2048
  int b = idx >> 11;
  int j = idx & (H2 - 1);
  dst_bf[idx] = f2bf(src[((size_t)b << 11) + ((j & 3) << 9) + (j >> 2)]);
}

// ---------------------------------------------------------------------------
// Host launcher
// ---------------------------------------------------------------------------
static inline void launch_gemm(hipStream_t st, int mode,
                               const unsigned short* A, int lda,
                               const unsigned short* Bw, const float* bias,
                               float* C, int ldc, unsigned short* Cbf,
                               const float* gate, int ldg,
                               float* out2, int ld2, int M, int N, int K) {
  int waves  = (M >> 4) * (N >> 6);
  int blocks = (waves + 3) >> 2;
  dim3 g(blocks), blk(128);
  switch (mode) {
    case 0:   // bias epilogue, f32 C only
      wmma_gemm_kernel<0, true, false><<<g, blk, 0, st>>>(
          A, lda, Bw, bias, C, ldc, Cbf, gate, ldg, out2, ld2, M, N, K);
      break;
    case 1:   // relu epilogue, bf16 Cbf only
      wmma_gemm_kernel<1, false, true><<<g, blk, 0, st>>>(
          A, lda, Bw, bias, C, ldc, Cbf, gate, ldg, out2, ld2, M, N, K);
      break;
    default:  // sigmoid-gate epilogue, f32 C + bf16 Cbf (+ optional out2)
      wmma_gemm_kernel<2, true, true><<<g, blk, 0, st>>>(
          A, lda, Bw, bias, C, ldc, Cbf, gate, ldg, out2, ld2, M, N, K);
      break;
  }
}

extern "C" void kernel_launch(void* const* d_in, const int* in_sizes, int n_in,
                              void* d_out, int out_size, void* d_ws,
                              size_t ws_size, hipStream_t stream) {
  const float* x    = (const float*)d_in[0];   // (B,S,H)
  const float* h0   = (const float*)d_in[1];   // (L,1,H)
  const float* Wih  = (const float*)d_in[2];   // (L,3H,H)
  const float* Whh  = (const float*)d_in[3];   // (L,3H,H)
  const float* bih  = (const float*)d_in[4];   // (L,3H)
  const float* bhh  = (const float*)d_in[5];   // (L,3H)
  const float* aw0  = (const float*)d_in[6];   // (L,H,H)
  const float* aw1  = (const float*)d_in[7];   // (L,GH,GH)
  const float* aw2  = (const float*)d_in[8];   // (L,GH,GH)
  const float* aw3  = (const float*)d_in[9];   // (L,H,2H)

  float* outp   = (float*)d_out;                         // (B,S,H)
  float* hstate = outp + (size_t)BB * SS * HH;           // (L,B,H) == hT slot

  // ---- workspace layout
  char* p = (char*)d_ws;
  size_t off = 0;
  auto carve = [&](size_t bytes) {
    void* q = p + off;
    off = (off + bytes + 255) & ~(size_t)255;
    return q;
  };
  const size_t nWih = (size_t)LL * H3 * HH;      // 12.58M
  const size_t nAw0 = (size_t)LL * HH * HH;      // 4.19M
  const size_t nAw1 = (size_t)LL * GHH * GHH;    // 1.05M
  const size_t nAw3 = (size_t)LL * HH * H2;      // 8.39M
  const size_t nX   = (size_t)BB * SS * HH;      // 16.78M
  unsigned short* wih_bf = (unsigned short*)carve(nWih * 2);
  unsigned short* whh_bf = (unsigned short*)carve(nWih * 2);
  unsigned short* aw0_bf = (unsigned short*)carve(nAw0 * 2);
  unsigned short* aw1_bf = (unsigned short*)carve(nAw1 * 2);
  unsigned short* aw2_bf = (unsigned short*)carve(nAw1 * 2);
  unsigned short* aw3_bf = (unsigned short*)carve(nAw3 * 2);
  unsigned short* x_bf   = (unsigned short*)carve(nX * 2);
  unsigned short* h_bf   = (unsigned short*)carve((size_t)LL * BB * HH * 2);
  unsigned short* se_bf  = (unsigned short*)carve((size_t)BB * H2 * 2);
  unsigned short* sh_bf  = (unsigned short*)carve((size_t)BB * H2 * 2);
  unsigned short* a2_bf  = (unsigned short*)carve((size_t)BB * H2 * 2);
  float* gi  = (float*)carve((size_t)BB * H3 * 4);
  float* gh  = (float*)carve((size_t)BB * H3 * 4);
  float* hpr = (float*)carve((size_t)BB * HH * 4);
  float* t0  = (float*)carve((size_t)BB * H2 * 4);
  (void)ws_size; (void)in_sizes; (void)n_in; (void)out_size;

  // ---- convert weights + x input fp32 -> bf16 (deterministic each call)
  auto conv = [&](const float* s, unsigned short* d, size_t n) {
    f32_to_bf16_kernel<<<dim3((unsigned)((n + 255) / 256)), dim3(256), 0,
                         stream>>>(s, d, (int)n);
  };
  conv(Wih, wih_bf, nWih);
  conv(Whh, whh_bf, nWih);
  conv(aw0, aw0_bf, nAw0);
  conv(aw1, aw1_bf, nAw1);
  conv(aw2, aw2_bf, nAw1);
  conv(aw3, aw3_bf, nAw3);
  conv(x, x_bf, nX);

  // ---- init hidden state (broadcast hidden0 over batch) + bf16 shadow
  init_h_kernel<<<dim3((LL * BB * HH) / 256), dim3(256), 0, stream>>>(
      h0, hstate, h_bf);

  const int nBH  = BB * HH;
  const int nB2H = BB * H2;

  for (int t = 0; t < SS; ++t) {
    for (int l = 0; l < LL; ++l) {
      const unsigned short* inA;
      int lda;
      if (l == 0) { inA = x_bf + (size_t)t * HH; lda = SS * HH; }
      else        { inA = h_bf + (size_t)(l - 1) * BB * HH; lda = HH; }
      float* hl = hstate + (size_t)l * BB * HH;
      unsigned short* hlbf = h_bf + (size_t)l * BB * HH;

      // GRU gates: gi = in @ Wih^T + bih ; gh = h @ Whh^T + bhh
      launch_gemm(stream, 0, inA, lda, wih_bf + (size_t)l * H3 * HH,
                  bih + (size_t)l * H3, gi, H3, nullptr, nullptr, 0,
                  nullptr, 0, BB, H3, HH);
      launch_gemm(stream, 0, hlbf, HH, whh_bf + (size_t)l * H3 * HH,
                  bhh + (size_t)l * H3, gh, H3, nullptr, nullptr, 0,
                  nullptr, 0, BB, H3, HH);
      gru_pointwise_kernel<<<dim3(nBH / 256), dim3(256), 0, stream>>>(
          gi, gh, hl, hpr);

      // SortAttn
      sort_rows_kernel<<<dim3(BB), dim3(512), 0, stream>>>(hpr, se_bf);
      launch_gemm(stream, 0, se_bf, HH, aw0_bf + (size_t)l * HH * HH, nullptr,
                  t0, HH, nullptr, nullptr, 0, nullptr, 0, 2 * BB, HH, HH);
      shuffle_kernel<<<dim3(nB2H / 256), dim3(256), 0, stream>>>(t0, sh_bf);
      launch_gemm(stream, 0, sh_bf, GHH, aw1_bf + (size_t)l * GHH * GHH,
                  nullptr, t0, GHH, nullptr, nullptr, 0, nullptr, 0,
                  4 * BB, GHH, GHH);
      shuffle_kernel<<<dim3(nB2H / 256), dim3(256), 0, stream>>>(t0, sh_bf);
      launch_gemm(stream, 1, sh_bf, GHH, aw2_bf + (size_t)l * GHH * GHH,
                  nullptr, nullptr, GHH, a2_bf, nullptr, 0, nullptr, 0,
                  4 * BB, GHH, GHH);
      // final linear + sigmoid gate -> new h (f32 + bf16 shadow), out[:,t,:]
      float* out2 = (l == LL - 1) ? (outp + (size_t)t * HH) : nullptr;
      launch_gemm(stream, 2, a2_bf, H2, aw3_bf + (size_t)l * HH * H2, nullptr,
                  hl, HH, hlbf, hpr, HH, out2, SS * HH, BB, HH, H2);
    }
  }
}
